// TensorEmbedding_58145267253391
// MI455X (gfx1250) — compile-verified, matrix operand order checked by reading, not run on hardware
//
#include <hip/hip_runtime.h>

#define HDIM 128
#define RBF  64
#define PI_F 3.14159265358979323846f

typedef __attribute__((ext_vector_type(16))) _Float16 v16h;
typedef __attribute__((ext_vector_type(8)))  float    v8f;

__device__ __forceinline__ float silu_f(float x) {
  return x * (1.0f / (1.0f + __expf(-x)));
}

// ---------------- kernel 0: zero accumulators ----------------
__global__ void k_zero(float* __restrict__ p, long n) {
  long i = (long)blockIdx.x * blockDim.x + threadIdx.x;
  if (i < n) p[i] = 0.0f;
}

// ---------------- kernel 1: node MLP -> Zh (f16) ----------------
// Z = silu(concat(emb[z], t) @ mix_w1 + b1) @ mix_w2 + b2
// Zt@w1 is split as Z@w1[0:H] + t*w1[H] (rank-1 term), so K stays 128.
__global__ void k_node_mlp(const int* __restrict__ z, const float* __restrict__ t,
                           const float* __restrict__ emb,
                           const float* __restrict__ w1, const float* __restrict__ b1,
                           const float* __restrict__ w2, const float* __restrict__ b2,
                           _Float16* __restrict__ Zh) {
  int n = blockIdx.x, h = threadIdx.x;
  __shared__ float zr[HDIM];
  __shared__ float hid[HDIM];
  zr[h] = emb[(long)z[n] * HDIM + h];
  float tv = t[n];
  __syncthreads();
  float acc = b1[h] + tv * w1[HDIM * HDIM + h];
  #pragma unroll 8
  for (int k = 0; k < HDIM; ++k) acc += zr[k] * w1[k * HDIM + h];
  hid[h] = silu_f(acc);
  __syncthreads();
  float acc2 = b2[h];
  #pragma unroll 8
  for (int k = 0; k < HDIM; ++k) acc2 += hid[k] * w2[k * HDIM + h];
  Zh[(long)n * HDIM + h] = (_Float16)acc2;
}

// ---------------- kernel 2: pack weights into WMMA B-fragment layout ----------
// Fragment (ks, nt) holds a 32xK=32, N=16 tile as 32 lanes x 16 f16 (contiguous
// per lane, so the edge kernel loads each lane's v16h with one b128-pair load).
// Lane mapping mirrors the documented 16-bit A layout with M<->N swapped:
//   n = nt*16 + (lane&15); khalf = lane>>4
//   i<8  -> k_local = i + khalf*8        (regs 0..3, K pairs)
//   i>=8 -> k_local = i + 8 + khalf*8    (regs 4..7)
__global__ void k_pack(const float* __restrict__ W, _Float16* __restrict__ dst, int nKsteps) {
  int tid = blockIdx.x * blockDim.x + threadIdx.x;
  int total = nKsteps * 8 * 512;
  if (tid >= total) return;
  int ks   = tid / (8 * 512);
  int rem  = tid % (8 * 512);
  int nt   = rem / 512;
  int li   = rem % 512;
  int lane = li >> 4;
  int i    = li & 15;
  int khalf = lane >> 4;
  int nn = nt * 16 + (lane & 15);
  int kb = (i < 8) ? (i + khalf * 8) : (i + 8 + khalf * 8);
  int k  = ks * 32 + kb;
  dst[tid] = (_Float16)W[k * HDIM + nn];
}

// ---------------- A-fragment builders (16-bit A, 16x32, documented layout) ---
__device__ __forceinline__ v16h zij_frag(const _Float16* __restrict__ Zh,
                                         int ks, int nodeS, int nodeD, int khalf) {
  union { v16h v; unsigned u[8]; } r;
  const _Float16* rowS = Zh + (long)nodeS * HDIM;
  const _Float16* rowD = Zh + (long)nodeD * HDIM;
  #pragma unroll
  for (int j = 0; j < 8; ++j) {
    int two = 2 * j;
    int kb = (two < 8) ? (two + khalf * 8) : (two + 8 + khalf * 8);
    int k = ks * 32 + kb;                       // K total = 256 = [Z[src] | Z[dst]]
    const _Float16* src = (k < HDIM) ? (rowS + k) : (rowD + (k - HDIM));
    r.u[j] = *(const unsigned*)src;             // 2 consecutive f16 per reg
  }
  return r.v;
}

__device__ __forceinline__ v16h attr_frag(const float* __restrict__ arow, int ks, int khalf) {
  union { v16h v; _Float16 h[16]; } r;
  #pragma unroll
  for (int j = 0; j < 8; ++j) {
    int two = 2 * j;
    int kb = (two < 8) ? (two + khalf * 8) : (two + 8 + khalf * 8);
    int k = ks * 32 + kb;
    float2 f = *(const float2*)(arow + k);
    r.h[2 * j]     = (_Float16)f.x;
    r.h[2 * j + 1] = (_Float16)f.y;
  }
  return r.v;
}

// ---------------- kernel 3: fused per-edge WMMA pipeline + scatter -----------
// Per wave: 16 edges. Zij = [16x256]@[256x128] via 8 k-steps of f16 WMMA,
// then C = (Zij+b)*cutoff, then fk = ([16x64]@[64x128]+b)*C for dp1..3,
// scattered as 1 (diag) + 3 (skew vec) + 6 (sym components) atomics.
__global__ void __launch_bounds__(256) k_edge(
    const int* __restrict__ ei, const float* __restrict__ ew,
    const float* __restrict__ evn, const float* __restrict__ eattr,
    const _Float16* __restrict__ Zh,
    const _Float16* __restrict__ packEmb2, const _Float16* __restrict__ packDp,
    const float* __restrict__ emb2_b,
    const float* __restrict__ dp1_b, const float* __restrict__ dp2_b,
    const float* __restrict__ dp3_b,
    float* __restrict__ accI, float* __restrict__ accA, float* __restrict__ accS,
    int E_) {
  // per-wave edge metadata: src, dst, cutoff, v0..2, sym 6 components
  __shared__ float meta[8][16][12];
  const int lane = threadIdx.x & 31;
  const int wv   = threadIdx.x >> 5;
  const long e0  = ((long)blockIdx.x * 8 + wv) * 16;

  if (lane < 16) {
    long e = e0 + lane;
    bool valid = (e < (long)E_);
    long ec = valid ? e : 0;
    int s = ei[ec];
    int d = ei[(long)E_ + ec];
    float w = ew[ec];
    float cut = (valid && (w < 5.0f)) ? 0.5f * (__cosf(w * (PI_F / 5.0f)) + 1.0f) : 0.0f;
    float v0 = evn[ec * 3 + 0], v1 = evn[ec * 3 + 1], v2 = evn[ec * 3 + 2];
    float tr3 = (v0 * v0 + v1 * v1 + v2 * v2) * (1.0f / 3.0f);
    float* mp = meta[wv][lane];
    mp[0] = __int_as_float(valid ? s : 0);
    mp[1] = __int_as_float(valid ? d : 0);
    mp[2] = cut;                                  // cut==0 neutralizes tail edges
    mp[3] = v0; mp[4] = v1; mp[5] = v2;
    mp[6] = v0 * v0 - tr3; mp[7]  = v0 * v1; mp[8]  = v0 * v2;
    mp[9] = v1 * v1 - tr3; mp[10] = v1 * v2; mp[11] = v2 * v2 - tr3;
  }
  __syncthreads();

  const int khalf = lane >> 4;
  const int nlo   = lane & 15;       // A row (edge) for this lane; also h%16 for C/D
  const int mBase = khalf * 8;       // C/D: reg r holds edge row mBase+r
  const int nodeS = __float_as_int(meta[wv][nlo][0]);
  const int nodeD = __float_as_int(meta[wv][nlo][1]);
  long ea = e0 + nlo; if (ea >= (long)E_) ea = (long)E_ - 1;
  const float* arow = eattr + ea * RBF;

  // shared A fragments for the three dp GEMMs (same edge_attr rows)
  v16h aAttr0 = attr_frag(arow, 0, khalf);
  v16h aAttr1 = attr_frag(arow, 1, khalf);

  const v16h* fE = (const v16h*)packEmb2;
  const v16h* fD = (const v16h*)packDp;
  const float* dpb[3] = { dp1_b, dp2_b, dp3_b };
  const v8f zv = {};

  for (int ntb = 0; ntb < 8; ntb += 4) {         // 2 chunks of 4 N-tiles (64 cols)
    // --- Zij GEMM: K = 256 (8 k-steps) ---
    v8f CZ[4] = { zv, zv, zv, zv };
    #pragma unroll
    for (int ks = 0; ks < 8; ++ks) {
      v16h a = zij_frag(Zh, ks, nodeS, nodeD, khalf);
      #pragma unroll
      for (int q = 0; q < 4; ++q) {
        v16h b = fE[(ks * 8 + ntb + q) * 32 + lane];
        CZ[q] = __builtin_amdgcn_wmma_f32_16x16x32_f16(
                    false, a, false, b, (short)0, CZ[q], false, false);
      }
    }
    // C = (Zij + emb2_b) * cutoff   (elementwise in WMMA C layout)
    #pragma unroll
    for (int q = 0; q < 4; ++q) {
      int h = (ntb + q) * 16 + nlo;
      float bb = emb2_b[h];
      #pragma unroll
      for (int r = 0; r < 8; ++r)
        CZ[q][r] = (CZ[q][r] + bb) * meta[wv][mBase + r][2];
    }
    // --- dp1..3 GEMMs (K=64) + structured scatter ---
    #pragma unroll
    for (int kk = 0; kk < 3; ++kk) {
      v8f F[4] = { zv, zv, zv, zv };
      #pragma unroll
      for (int q = 0; q < 4; ++q) {
        v16h b0 = fD[((kk * 2 + 0) * 8 + ntb + q) * 32 + lane];
        F[q] = __builtin_amdgcn_wmma_f32_16x16x32_f16(
                   false, aAttr0, false, b0, (short)0, F[q], false, false);
        v16h b1 = fD[((kk * 2 + 1) * 8 + ntb + q) * 32 + lane];
        F[q] = __builtin_amdgcn_wmma_f32_16x16x32_f16(
                   false, aAttr1, false, b1, (short)0, F[q], false, false);
      }
      #pragma unroll
      for (int q = 0; q < 4; ++q) {
        int h = (ntb + q) * 16 + nlo;
        float bb = dpb[kk][h];
        #pragma unroll
        for (int r = 0; r < 8; ++r) {
          int m = mBase + r;
          float f = (F[q][r] + bb) * CZ[q][r];
          int node = __float_as_int(meta[wv][m][0]);   // scatter at src
          long base = (long)node * HDIM + h;
          if (kk == 0) {
            atomicAdd(accI + base, f);                  // Iij = f1 * eye
          } else if (kk == 1) {
            atomicAdd(accA + base * 3 + 0, f * meta[wv][m][3]);  // skew(sum f2*v)
            atomicAdd(accA + base * 3 + 1, f * meta[wv][m][4]);
            atomicAdd(accA + base * 3 + 2, f * meta[wv][m][5]);
          } else {
            atomicAdd(accS + base * 6 + 0, f * meta[wv][m][6]);  // sym components
            atomicAdd(accS + base * 6 + 1, f * meta[wv][m][7]);
            atomicAdd(accS + base * 6 + 2, f * meta[wv][m][8]);
            atomicAdd(accS + base * 6 + 3, f * meta[wv][m][9]);
            atomicAdd(accS + base * 6 + 4, f * meta[wv][m][10]);
            atomicAdd(accS + base * 6 + 5, f * meta[wv][m][11]);
          }
        }
      }
    }
  }
}

// ---------------- kernel 4: norm branch (||T||^2 -> LN -> 2 MLPs) ------------
__global__ void k_norm(const float* __restrict__ accI, const float* __restrict__ accA,
                       const float* __restrict__ accS,
                       const float* __restrict__ ln_g, const float* __restrict__ ln_b,
                       const float* __restrict__ ls1_w, const float* __restrict__ ls1_b,
                       const float* __restrict__ ls2_w, const float* __restrict__ ls2_b,
                       float* __restrict__ nv) {
  int n = blockIdx.x, h = threadIdx.x;
  __shared__ float y[HDIM];
  __shared__ float hid[2 * HDIM];
  __shared__ float red[HDIM];
  __shared__ float red2[HDIM];
  long base = (long)n * HDIM + h;
  float d  = accI[base];
  float a0 = accA[base * 3 + 0], a1 = accA[base * 3 + 1], a2 = accA[base * 3 + 2];
  float s0 = accS[base * 6 + 0], s1 = accS[base * 6 + 1], s2 = accS[base * 6 + 2];
  float s3 = accS[base * 6 + 3], s4 = accS[base * 6 + 4], s5 = accS[base * 6 + 5];
  // T = d*I + skew(a) + S ; Frobenius^2
  float t00 = d + s0, t11 = d + s3, t22 = d + s5;
  float t01 = s1 - a2, t10 = s1 + a2;
  float t02 = s2 + a1, t20 = s2 - a1;
  float t12 = s4 - a0, t21 = s4 + a0;
  float nn = t00*t00 + t01*t01 + t02*t02 + t10*t10 + t11*t11 + t12*t12
           + t20*t20 + t21*t21 + t22*t22;
  nn = fmaxf(nn, 0.01f);
  red[h] = nn; red2[h] = nn * nn;
  __syncthreads();
  #pragma unroll
  for (int s = HDIM / 2; s > 0; s >>= 1) {
    if (h < s) { red[h] += red[h + s]; red2[h] += red2[h + s]; }
    __syncthreads();
  }
  float mu  = red[0] * (1.0f / HDIM);
  float var = red2[0] * (1.0f / HDIM) - mu * mu;
  y[h] = (nn - mu) * rsqrtf(var + 1e-5f) * ln_g[h] + ln_b[h];
  __syncthreads();
  #pragma unroll
  for (int rep = 0; rep < 2; ++rep) {
    int j = h + rep * HDIM;
    float acc = ls1_b[j];
    #pragma unroll 8
    for (int k = 0; k < HDIM; ++k) acc += y[k] * ls1_w[k * (2 * HDIM) + j];
    hid[j] = silu_f(acc);
  }
  __syncthreads();
  #pragma unroll
  for (int c = 0; c < 3; ++c) {
    int j = h * 3 + c;
    float acc = ls2_b[j];
    #pragma unroll 8
    for (int k = 0; k < 2 * HDIM; ++k) acc += hid[k] * ls2_w[k * (3 * HDIM) + j];
    nv[(long)n * (3 * HDIM) + j] = silu_f(acc);
  }
}

// ---------------- kernel 5: channel mix + tensor reconstruction --------------
__global__ void k_out(const float* __restrict__ accI, const float* __restrict__ accA,
                      const float* __restrict__ accS,
                      const float* __restrict__ lt0, const float* __restrict__ lt1,
                      const float* __restrict__ lt2,
                      const float* __restrict__ nv, float* __restrict__ out) {
  int n = blockIdx.x, k = threadIdx.x;
  __shared__ float sI[HDIM];
  __shared__ float sA[3][HDIM];
  __shared__ float sS[6][HDIM];
  long base = (long)n * HDIM + k;
  sI[k] = accI[base];
  #pragma unroll
  for (int c = 0; c < 3; ++c) sA[c][k] = accA[base * 3 + c];
  #pragma unroll
  for (int c = 0; c < 6; ++c) sS[c][k] = accS[base * 6 + c];
  __syncthreads();
  float im = 0.f, a0 = 0.f, a1 = 0.f, a2 = 0.f;
  float s0 = 0.f, s1 = 0.f, s2 = 0.f, s3 = 0.f, s4 = 0.f, s5 = 0.f;
  #pragma unroll 4
  for (int hh = 0; hh < HDIM; ++hh) {
    float w0 = lt0[hh * HDIM + k];
    float w1 = lt1[hh * HDIM + k];
    float w2 = lt2[hh * HDIM + k];
    im += sI[hh] * w0;
    a0 += sA[0][hh] * w1; a1 += sA[1][hh] * w1; a2 += sA[2][hh] * w1;
    s0 += sS[0][hh] * w2; s1 += sS[1][hh] * w2; s2 += sS[2][hh] * w2;
    s3 += sS[3][hh] * w2; s4 += sS[4][hh] * w2; s5 += sS[5][hh] * w2;
  }
  float n0 = nv[(long)n * 384 + k * 3 + 0];
  float n1 = nv[(long)n * 384 + k * 3 + 1];
  float n2 = nv[(long)n * 384 + k * 3 + 2];
  float* o = out + base * 9;
  o[0] =  im * n0 + s0 * n2;
  o[1] = -a2 * n1 + s1 * n2;
  o[2] =  a1 * n1 + s2 * n2;
  o[3] =  a2 * n1 + s1 * n2;
  o[4] =  im * n0 + s3 * n2;
  o[5] = -a0 * n1 + s4 * n2;
  o[6] = -a1 * n1 + s2 * n2;
  o[7] =  a0 * n1 + s4 * n2;
  o[8] =  im * n0 + s5 * n2;
}

// ---------------- host launcher ----------------
extern "C" void kernel_launch(void* const* d_in, const int* in_sizes, int n_in,
                              void* d_out, int out_size, void* d_ws, size_t ws_size,
                              hipStream_t stream) {
  (void)n_in; (void)out_size; (void)ws_size;
  const int*   z      = (const int*)  d_in[0];
  const float* t      = (const float*)d_in[1];
  const int*   ei     = (const int*)  d_in[2];
  const float* ew     = (const float*)d_in[3];
  const float* evn    = (const float*)d_in[4];
  const float* eattr  = (const float*)d_in[5];
  const float* emb    = (const float*)d_in[6];
  const float* mix_w1 = (const float*)d_in[7];
  const float* mix_b1 = (const float*)d_in[8];
  const float* mix_w2 = (const float*)d_in[9];
  const float* mix_b2 = (const float*)d_in[10];
  const float* emb2_w = (const float*)d_in[11];
  const float* emb2_b = (const float*)d_in[12];
  const float* dp1_w  = (const float*)d_in[13];
  const float* dp1_b  = (const float*)d_in[14];
  const float* dp2_w  = (const float*)d_in[15];
  const float* dp2_b  = (const float*)d_in[16];
  const float* dp3_w  = (const float*)d_in[17];
  const float* dp3_b  = (const float*)d_in[18];
  const float* lt0    = (const float*)d_in[19];
  const float* lt1    = (const float*)d_in[20];
  const float* lt2    = (const float*)d_in[21];
  const float* ls1_w  = (const float*)d_in[22];
  const float* ls1_b  = (const float*)d_in[23];
  const float* ls2_w  = (const float*)d_in[24];
  const float* ls2_b  = (const float*)d_in[25];
  const float* ln_g   = (const float*)d_in[26];
  const float* ln_b   = (const float*)d_in[27];

  const int N_ = in_sizes[0];      // z: (N,)
  const int E_ = in_sizes[3];      // edge_weight: (E,)

  // workspace layout (all 256B-aligned given N multiple of 16)
  float* accI = (float*)d_ws;                         // N*H
  float* accA = accI + (size_t)N_ * HDIM;             // N*H*3
  float* accS = accA + (size_t)N_ * HDIM * 3;         // N*H*6
  float* nv   = accS + (size_t)N_ * HDIM * 6;         // N*3H
  _Float16* Zh       = (_Float16*)(nv + (size_t)N_ * 3 * HDIM);  // N*H f16
  _Float16* packEmb2 = Zh + (size_t)N_ * HDIM;        // 8*8*512 f16
  _Float16* packDp   = packEmb2 + 8 * 8 * 512;        // 3*2*8*512 f16

  long zn = (long)N_ * HDIM * 10;                     // accI|accA|accS contiguous
  k_zero<<<(int)((zn + 1023) / 1024), 1024, 0, stream>>>(accI, zn);

  k_node_mlp<<<N_, HDIM, 0, stream>>>(z, t, emb, mix_w1, mix_b1, mix_w2, mix_b2, Zh);

  k_pack<<<(8 * 8 * 512 + 255) / 256, 256, 0, stream>>>(emb2_w, packEmb2, 8);
  k_pack<<<(2 * 8 * 512 + 255) / 256, 256, 0, stream>>>(dp1_w, packDp + 0 * 2 * 8 * 512, 2);
  k_pack<<<(2 * 8 * 512 + 255) / 256, 256, 0, stream>>>(dp2_w, packDp + 1 * 2 * 8 * 512, 2);
  k_pack<<<(2 * 8 * 512 + 255) / 256, 256, 0, stream>>>(dp3_w, packDp + 2 * 2 * 8 * 512, 2);

  int tiles  = (E_ + 15) / 16;        // 16 edges per wave
  int blocks = (tiles + 7) / 8;       // 8 waves per block
  k_edge<<<blocks, 256, 0, stream>>>(ei, ew, evn, eattr, Zh, packEmb2, packDp,
                                     emb2_b, dp1_b, dp2_b, dp3_b,
                                     accI, accA, accS, E_);

  k_norm<<<N_, HDIM, 0, stream>>>(accI, accA, accS, ln_g, ln_b,
                                  ls1_w, ls1_b, ls2_w, ls2_b, nv);

  k_out<<<N_, HDIM, 0, stream>>>(accI, accA, accS, lt0, lt1, lt2, nv, (float*)d_out);
}
